// SparseTransformerCrossBlock_45689862095237
// MI455X (gfx1250) — compile-verified, matrix operand order checked by reading, not run on hardware
//
#include <hip/hip_runtime.h>
#include <cstdint>
#include <cstddef>

typedef unsigned short u16;
typedef __attribute__((ext_vector_type(16))) __bf16 v16bf;
typedef __attribute__((ext_vector_type(8)))  float  v8f;

struct alignas(16) I4 { int x, y, z, w; };
struct alignas(16) AB32 { I4 lo, hi; };     // one WMMA A/B fragment: 32 bytes
struct alignas(16) P32 { u16 s[16]; };      // packed bf16 P fragment

__device__ __forceinline__ u16 f2bf(float f) {
  unsigned u = __float_as_uint(f);
  u += 0x7FFFu + ((u >> 16) & 1u);          // round-to-nearest-even
  return (u16)(u >> 16);
}

__device__ __forceinline__ v8f vzero8() {
  v8f z;
#pragma unroll
  for (int i = 0; i < 8; i++) z[i] = 0.f;
  return z;
}

__device__ __forceinline__ v8f wmma_bf16(const AB32& a, const AB32& b, v8f c) {
  return __builtin_amdgcn_wmma_f32_16x16x32_bf16(
      false, __builtin_bit_cast(v16bf, a),
      false, __builtin_bit_cast(v16bf, b),
      (short)0, c, false, false);
}

// ---------------------------------------------------------------------------
// Tiled weight transpose + fp32 -> bf16:  W[K][Nout] -> Wt[Nout][K]
// 32x32 tile through LDS so both global reads and writes are contiguous.
// grid = (Nout/32, K/32), block = 256
// ---------------------------------------------------------------------------
__launch_bounds__(256)
__global__ void wtr_k(const float* __restrict__ W, u16* __restrict__ Wt,
                      int K, int Nout) {
  __shared__ u16 t[32 * 36];
  const int n0 = blockIdx.x * 32;
  const int k0 = blockIdx.y * 32;
  const int c  = threadIdx.x & 31;
  const int r0 = threadIdx.x >> 5;      // 0..7
#pragma unroll
  for (int i = 0; i < 4; i++) {
    int r = r0 + 8 * i;                 // k index
    t[c * 36 + r] = f2bf(W[(size_t)(k0 + r) * Nout + n0 + c]);
  }
  __syncthreads();
#pragma unroll
  for (int i = 0; i < 4; i++) {
    int r = r0 + 8 * i;                 // n index
    Wt[(size_t)(n0 + r) * K + k0 + c] = t[r * 36 + c];
  }
}

// fp32 -> bf16 elementwise
__global__ void cvt_k(const float* __restrict__ X, u16* __restrict__ Y, long total) {
  long i = (long)blockIdx.x * blockDim.x + threadIdx.x;
  if (i < total) Y[i] = f2bf(X[i]);
}

// ---------------------------------------------------------------------------
// LayerNorm over C=1024: one block per row, fp32 in -> bf16 out
// ---------------------------------------------------------------------------
__launch_bounds__(256)
__global__ void ln_k(const float* __restrict__ X, u16* __restrict__ Hout, int Crow) {
  const int row = blockIdx.x;
  const int tid = threadIdx.x;
  const float* xr = X + (size_t)row * Crow;
  float vals[4];
  float s = 0.f, s2 = 0.f;
#pragma unroll
  for (int i = 0; i < 4; i++) {
    vals[i] = xr[tid + 256 * i];
    s  += vals[i];
    s2 += vals[i] * vals[i];
  }
#pragma unroll
  for (int off = 16; off >= 1; off >>= 1) {
    s  += __shfl_xor(s,  off, 32);
    s2 += __shfl_xor(s2, off, 32);
  }
  __shared__ float ss[8], ss2[8];
  if ((tid & 31) == 0) { ss[tid >> 5] = s; ss2[tid >> 5] = s2; }
  __syncthreads();
  float ts = 0.f, ts2 = 0.f;
#pragma unroll
  for (int i = 0; i < 8; i++) { ts += ss[i]; ts2 += ss2[i]; }
  float mu   = ts / (float)Crow;
  float var  = ts2 / (float)Crow - mu * mu;
  float rstd = rsqrtf(var + 1e-6f);
  u16* hr = Hout + (size_t)row * Crow;
#pragma unroll
  for (int i = 0; i < 4; i++) hr[tid + 256 * i] = f2bf((vals[i] - mu) * rstd);
}

// ---------------------------------------------------------------------------
// bf16 GEMM: out[M][N] = A[M][K] @ Wt[N][K]^T + bias, templated epilogue.
// Block tile 128x256, 256 threads (8 waves, 2x4), wave tile 64x64.
// Register-prefetch double buffering: next k-step's global loads are issued
// before the 16-WMMA compute phase so they overlap it.
// ---------------------------------------------------------------------------
#define EPI_BF16 0
#define EPI_GELU 1
#define EPI_RES  2

template <int EPI>
__launch_bounds__(256)
__global__ void gemm_bf16_k(const u16* __restrict__ A, const u16* __restrict__ Wt,
                            const float* __restrict__ bias, const float* res,
                            void* outp, int Nout, int K) {
  __shared__ u16 As[128 * 40];   // 128 rows x 32 cols, pad to 40
  __shared__ u16 Bs[256 * 40];   // 256 rows x 32 cols, pad to 40
  const int tid  = threadIdx.x;
  const int lane = tid & 31;
  const int w    = tid >> 5;
  const int wm   = (w & 1) * 64;   // wave M offset within block tile
  const int wn   = (w >> 1) * 64;  // wave N offset
  const int m0   = blockIdx.y * 128;
  const int n0   = blockIdx.x * 256;
  const int l16  = lane & 15;
  const int lh   = lane >> 4;      // lane half
  const int kb   = lh * 8;         // K base within fragment

  v8f c[4][4];
#pragma unroll
  for (int i = 0; i < 4; i++)
#pragma unroll
    for (int j = 0; j < 4; j++) c[i][j] = vzero8();

  // global->LDS loader assignment
  const int arow = tid >> 1;            // 0..127
  const int acb  = (tid & 1) * 16;      // 32B half-row of A
  const int brow = tid;                 // 0..255, full 64B row of B

  I4 pa0, pa1, pb0, pb1, pb2, pb3;
  const u16* aP = &A [(size_t)(m0 + arow) * K + acb];
  const u16* bP = &Wt[(size_t)(n0 + brow) * K];

  // prologue prefetch (k0 = 0)
  pa0 = *(const I4*)(aP);      pa1 = *(const I4*)(aP + 8);
  pb0 = *(const I4*)(bP);      pb1 = *(const I4*)(bP + 8);
  pb2 = *(const I4*)(bP + 16); pb3 = *(const I4*)(bP + 24);

  for (int k0 = 0; k0 < K; k0 += 32) {
    *(I4*)&As[arow * 40 + acb]     = pa0;
    *(I4*)&As[arow * 40 + acb + 8] = pa1;
    *(I4*)&Bs[brow * 40 +  0] = pb0;
    *(I4*)&Bs[brow * 40 +  8] = pb1;
    *(I4*)&Bs[brow * 40 + 16] = pb2;
    *(I4*)&Bs[brow * 40 + 24] = pb3;
    __syncthreads();

    if (k0 + 32 < K) {   // prefetch next tile; overlaps WMMA phase below
      const u16* ap = aP + k0 + 32;
      const u16* bp = bP + k0 + 32;
      pa0 = *(const I4*)(ap);      pa1 = *(const I4*)(ap + 8);
      pb0 = *(const I4*)(bp);      pb1 = *(const I4*)(bp + 8);
      pb2 = *(const I4*)(bp + 16); pb3 = *(const I4*)(bp + 24);
    }

    AB32 a[4], b[4];
#pragma unroll
    for (int i = 0; i < 4; i++) {
      int row = wm + 16 * i + l16;
      a[i].lo = *(const I4*)&As[row * 40 + kb];
      a[i].hi = *(const I4*)&As[row * 40 + 16 + kb];
    }
#pragma unroll
    for (int j = 0; j < 4; j++) {
      int row = wn + 16 * j + l16;
      b[j].lo = *(const I4*)&Bs[row * 40 + kb];
      b[j].hi = *(const I4*)&Bs[row * 40 + 16 + kb];
    }
#pragma unroll
    for (int i = 0; i < 4; i++)
#pragma unroll
      for (int j = 0; j < 4; j++)
        c[i][j] = wmma_bf16(a[i], b[j], c[i][j]);
    __syncthreads();
  }

  // Epilogue: C layout -> lane l holds col n = l16, rows m = v + 8*lh
#pragma unroll
  for (int i = 0; i < 4; i++) {
#pragma unroll
    for (int j = 0; j < 4; j++) {
      int n = n0 + wn + 16 * j + l16;
      float bv = bias[n];
#pragma unroll
      for (int v = 0; v < 8; v++) {
        int m = m0 + wm + 16 * i + v + 8 * lh;
        float val = c[i][j][v] + bv;
        size_t idx = (size_t)m * Nout + n;
        if (EPI == EPI_BF16) {
          ((u16*)outp)[idx] = f2bf(val);
        } else if (EPI == EPI_GELU) {
          float t = 0.7978845608028654f * (val + 0.044715f * val * val * val);
          float g = 0.5f * val * (1.f + tanhf(t));
          ((u16*)outp)[idx] = f2bf(g);
        } else {
          ((float*)outp)[idx] = res[idx] + val;
        }
      }
    }
  }
}

// ---------------------------------------------------------------------------
// Flash attention (bf16 QKV, fp32 softmax/accum).
// Computes S^T = K·Q^T so softmax stats are per-lane; converts P^T in-lane
// into the B-fragment for O^T = V^T·P^T.  Block = 4 waves, 64 queries.
// ---------------------------------------------------------------------------
__launch_bounds__(128)
__global__ void flash_k(const u16* __restrict__ Qb, const u16* __restrict__ Kb,
                        const u16* __restrict__ Vb, u16* __restrict__ Out,
                        int nQ, int nK, int qStride, int kStride,
                        int kOffBase, int vOffBase, float scale) {
  __shared__ u16 Ks[32 * 72];   // 32 keys x 64 d, pad to 72
  __shared__ u16 Vt[64 * 40];   // transposed: 64 d x 32 keys, pad to 40
  const int b    = blockIdx.z;
  const int h    = blockIdx.y;
  const int tid  = threadIdx.x;
  const int w    = tid >> 5;
  const int lane = tid & 31;
  const int l16  = lane & 15;
  const int lh   = lane >> 4;
  const int kb8  = lh * 8;

  const int qRow  = b * nQ + blockIdx.x * 64 + w * 16 + l16; // this lane's query
  const int kRow0 = b * nK;
  const int qOff  = h * 64;
  const int kOff  = kOffBase + h * 64;
  const int vOff  = vOffBase + h * 64;

  // Loop-invariant Q fragments (B operand of S^T WMMA): col = q = l16
  AB32 qf[2];
#pragma unroll
  for (int dk = 0; dk < 2; dk++) {
    const u16* qp = Qb + (size_t)qRow * qStride + qOff + dk * 32 + kb8;
    qf[dk].lo = *(const I4*)qp;
    qf[dk].hi = *(const I4*)(qp + 16);
  }

  v8f o[4];
#pragma unroll
  for (int dt = 0; dt < 4; dt++) o[dt] = vzero8();
  float mprev = -1e30f, lsum = 0.f;

  const int r  = tid >> 2;        // 0..31: key row within chunk
  const int cg = (tid & 3) * 8;   // d column group

  for (int kc = 0; kc < nK; kc += 32) {
    // stage K chunk row-major
    *(I4*)&Ks[r * 72 + cg] =
        *(const I4*)&Kb[(size_t)(kRow0 + kc + r) * kStride + kOff + cg];
    // stage V chunk transposed (Vt[d][key])
    I4 vv = *(const I4*)&Vb[(size_t)(kRow0 + kc + r) * kStride + vOff + cg];
    const u16* pv = (const u16*)&vv;
#pragma unroll
    for (int ii = 0; ii < 8; ii++) Vt[(cg + ii) * 40 + r] = pv[ii];
    __syncthreads();

    // S^T = K x Q^T  (two 16-key subtiles, K-dim = d = 64 -> 2 steps)
    v8f s0 = vzero8(), s1 = vzero8();
#pragma unroll
    for (int dk = 0; dk < 2; dk++) {
      AB32 ka;
      int row = l16;
      ka.lo = *(const I4*)&Ks[row * 72 + dk * 32 + kb8];
      ka.hi = *(const I4*)&Ks[row * 72 + dk * 32 + 16 + kb8];
      s0 = wmma_bf16(ka, qf[dk], s0);
      row = 16 + l16;
      ka.lo = *(const I4*)&Ks[row * 72 + dk * 32 + kb8];
      ka.hi = *(const I4*)&Ks[row * 72 + dk * 32 + 16 + kb8];
      s1 = wmma_bf16(ka, qf[dk], s1);
    }

    // Online softmax: lane holds 16 scores of its q-column; partner lane
    // (xor 16) holds the complementary 16 of the 32-key chunk.
    float p0[8], p1[8];
    float cm = -1e30f;
#pragma unroll
    for (int v = 0; v < 8; v++) {
      p0[v] = s0[v] * scale;
      p1[v] = s1[v] * scale;
      cm = fmaxf(cm, fmaxf(p0[v], p1[v]));
    }
    cm = fmaxf(cm, __shfl_xor(cm, 16, 32));
    float mnew = fmaxf(mprev, cm);
    float rsc  = __expf(mprev - mnew);
    float csum = 0.f;
    P32 pf;
#pragma unroll
    for (int v = 0; v < 8; v++) {
      p0[v] = __expf(p0[v] - mnew);
      p1[v] = __expf(p1[v] - mnew);
      csum += p0[v] + p1[v];
      pf.s[v]     = f2bf(p0[v]);   // S^T C-layout row v  == B-frag elem v
      pf.s[8 + v] = f2bf(p1[v]);   // keys 16..31 subtile == B-frag elems 8..15
    }
    csum += __shfl_xor(csum, 16, 32);
    lsum  = lsum * rsc + csum;
    mprev = mnew;
#pragma unroll
    for (int dt = 0; dt < 4; dt++)
#pragma unroll
      for (int v = 0; v < 8; v++) o[dt][v] *= rsc;

    // O^T += V^T x P^T   (A = V^T from transposed LDS tile)
    AB32 pb = __builtin_bit_cast(AB32, pf);
#pragma unroll
    for (int dt = 0; dt < 4; dt++) {
      AB32 va;
      int d = dt * 16 + l16;
      va.lo = *(const I4*)&Vt[d * 40 + kb8];
      va.hi = *(const I4*)&Vt[d * 40 + 16 + kb8];
      o[dt] = wmma_bf16(va, pb, o[dt]);
    }
    __syncthreads();
  }

  // O^T layout: col = q = l16 (matches this lane's qRow), row = d = v + 8*lh
  float inv = 1.f / lsum;
  u16* orow = Out + (size_t)qRow * 1024 + h * 64;
#pragma unroll
  for (int dt = 0; dt < 4; dt++)
#pragma unroll
    for (int v = 0; v < 8; v++) {
      int d = dt * 16 + v + 8 * lh;
      orow[d] = f2bf(o[dt][v] * inv);
    }
}

// ---------------------------------------------------------------------------
// Host-side pipeline
// ---------------------------------------------------------------------------
extern "C" void kernel_launch(void* const* d_in, const int* in_sizes, int n_in,
                              void* d_out, int out_size, void* d_ws, size_t ws_size,
                              hipStream_t stream) {
  (void)in_sizes; (void)n_in; (void)out_size; (void)ws_size;
  const float* x    = (const float*)d_in[0];
  const float* ctx  = (const float*)d_in[1];
  const float* Wqkv = (const float*)d_in[2];  const float* bqkv = (const float*)d_in[3];
  const float* Wos  = (const float*)d_in[4];  const float* bos  = (const float*)d_in[5];
  const float* Wq   = (const float*)d_in[6];  const float* bq   = (const float*)d_in[7];
  const float* Wkv  = (const float*)d_in[8];  const float* bkv  = (const float*)d_in[9];
  const float* Woc  = (const float*)d_in[10]; const float* boc  = (const float*)d_in[11];
  const float* W1   = (const float*)d_in[12]; const float* b1   = (const float*)d_in[13];
  const float* W2   = (const float*)d_in[14]; const float* b2   = (const float*)d_in[15];
  float* out = (float*)d_out;   // fp32 residual stream lives here

  char* p = (char*)d_ws;
  auto alloc = [&](size_t elems) { u16* q = (u16*)p; p += elems * sizeof(u16); return q; };
  u16* wt_qkv = alloc((size_t)3072 * 1024);
  u16* wt_os  = alloc((size_t)1024 * 1024);
  u16* wt_q   = alloc((size_t)1024 * 1024);
  u16* wt_kv  = alloc((size_t)2048 * 1024);
  u16* wt_oc  = alloc((size_t)1024 * 1024);
  u16* wt_1   = alloc((size_t)4096 * 1024);
  u16* wt_2   = alloc((size_t)1024 * 4096);
  u16* hbuf   = alloc((size_t)4096 * 1024);
  u16* qkv    = alloc((size_t)4096 * 3072);
  u16* attn   = alloc((size_t)4096 * 1024);
  u16* qb     = alloc((size_t)4096 * 1024);
  u16* kvb    = alloc((size_t)2048 * 2048);
  u16* ctxb   = alloc((size_t)2048 * 1024);
  u16* ffn1   = alloc((size_t)4096 * 4096);

  auto tr = [&](const float* W, u16* Wt, int K, int Nout) {
    wtr_k<<<dim3(Nout / 32, K / 32), 256, 0, stream>>>(W, Wt, K, Nout);
  };
  tr(Wqkv, wt_qkv, 1024, 3072);
  tr(Wos,  wt_os,  1024, 1024);
  tr(Wq,   wt_q,   1024, 1024);
  tr(Wkv,  wt_kv,  1024, 2048);
  tr(Woc,  wt_oc,  1024, 1024);
  tr(W1,   wt_1,   1024, 4096);
  tr(W2,   wt_2,   4096, 1024);
  cvt_k<<<dim3((2048 * 1024) / 256), 256, 0, stream>>>(ctx, ctxb, (long)2048 * 1024);

  // --- self-attention sub-block ---
  ln_k<<<4096, 256, 0, stream>>>(x, hbuf, 1024);
  gemm_bf16_k<EPI_BF16><<<dim3(12, 32), 256, 0, stream>>>(
      hbuf, wt_qkv, bqkv, nullptr, qkv, 3072, 1024);
  // qkv row layout: [q(0..1023) | k(1024..2047) | v(2048..3071)], head = h*64
  flash_k<<<dim3(32, 16, 2), 128, 0, stream>>>(
      qkv, qkv, qkv, attn, 2048, 2048, 3072, 3072, 1024, 2048, 0.125f);
  gemm_bf16_k<EPI_RES><<<dim3(4, 32), 256, 0, stream>>>(
      attn, wt_os, bos, x, out, 1024, 1024);

  // --- cross-attention sub-block ---
  ln_k<<<4096, 256, 0, stream>>>(out, hbuf, 1024);
  gemm_bf16_k<EPI_BF16><<<dim3(4, 32), 256, 0, stream>>>(
      hbuf, wt_q, bq, nullptr, qb, 1024, 1024);
  gemm_bf16_k<EPI_BF16><<<dim3(8, 16), 256, 0, stream>>>(
      ctxb, wt_kv, bkv, nullptr, kvb, 2048, 1024);
  flash_k<<<dim3(32, 16, 2), 128, 0, stream>>>(
      qb, kvb, kvb, attn, 2048, 1024, 1024, 2048, 0, 1024, 0.125f);
  gemm_bf16_k<EPI_RES><<<dim3(4, 32), 256, 0, stream>>>(
      attn, wt_oc, boc, out, out, 1024, 1024);

  // --- MLP sub-block ---
  ln_k<<<4096, 256, 0, stream>>>(out, hbuf, 1024);
  gemm_bf16_k<EPI_GELU><<<dim3(16, 32), 256, 0, stream>>>(
      hbuf, wt_1, b1, nullptr, ffn1, 4096, 1024);
  gemm_bf16_k<EPI_RES><<<dim3(4, 32), 256, 0, stream>>>(
      ffn1, wt_2, b2, out, out, 1024, 4096);
}
